// GAT_KH_18013092839768
// MI455X (gfx1250) — compile-verified, matrix operand order checked by reading, not run on hardware
//
#include <hip/hip_runtime.h>
#include <hip/hip_bf16.h>
#include <math.h>
#include <stdint.h>

// ---------------------------------------------------------------------------
// GAT k-hop network on MI455X (gfx1250, wave32, WMMA).
// GEMMs (13x [50000x256]x[256x256]) run on v_wmma_f32_16x16x32_bf16, staged
// through LDS with gfx1250 async global->LDS DMA (ASYNCcnt), double-buffered.
// Edge softmax/scatter uses hardware f32 atomics (output is L2-resident).
// ---------------------------------------------------------------------------

#define NNODES 50000
#define NEDGES 300000
#define DMODEL 256
#define NHEADS 8
#define DHEAD  32
#define MPAD   50048   // NNODES padded to multiple of 128 (zero rows)

typedef __attribute__((ext_vector_type(16))) __bf16 v16bf;
typedef __attribute__((ext_vector_type(8)))  float  v8f;

// ---------------------------------------------------------------------------
// fp32 -> bf16 converters (one streaming pass; lets the GEMM stage pure bf16
// tiles with async DMA, no VALU conversion, no bounds guards).
// ---------------------------------------------------------------------------
__global__ void cvtA_kernel(const float* __restrict__ in, __bf16* __restrict__ outp, int M)
{
  int t = blockIdx.x * blockDim.x + threadIdx.x;
  if (t >= MPAD * DMODEL) return;
  int row = t >> 8;
  outp[t] = (__bf16)((row < M) ? in[t] : 0.0f);
}

// Wt[n*256 + k] = W[k*256 + n]  (bf16, K contiguous per output column)
__global__ void cvtW_kernel(const float* __restrict__ W, __bf16* __restrict__ Wt)
{
  int t = blockIdx.x * blockDim.x + threadIdx.x;
  if (t >= DMODEL * DMODEL) return;
  int n = t >> 8, k = t & 255;
  Wt[t] = (__bf16)W[(size_t)k * DMODEL + n];
}

// ---------------------------------------------------------------------------
// WMMA GEMM: C[M,256] = act( A[M,256] @ W + bias )        (mode==0)
//            Acc[M,256] += decay * act( A @ W + bias )    (mode==1)
// Block: 256 threads (8 waves). Block tile: 128 rows x 256 cols.
// Wave w owns cols [32w,32w+32): 8x2 WMMA 16x16 tiles, K-loop of 8 x 32.
// Tiles staged via global_load_async_to_lds_b128, double-buffered (48KB LDS).
// ---------------------------------------------------------------------------
__device__ __forceinline__ void async_cp16(unsigned lds_off, const void* g)
{
  asm volatile("global_load_async_to_lds_b128 %0, %1, off"
               :: "v"(lds_off), "v"((unsigned long long)(uintptr_t)g)
               : "memory");
}

__global__ __launch_bounds__(256) void gemm_wmma_async(
    const __bf16* __restrict__ Abf, const __bf16* __restrict__ Wt,
    const float* __restrict__ bias, float* __restrict__ Cout,
    float* __restrict__ Acc, int M, float slope, int use_act, int use_bias,
    int mode, float decay)
{
  __shared__ __bf16 ldsA[2][128 * 32];   // 2 x 8KB
  __shared__ __bf16 ldsB[2][256 * 32];   // 2 x 16KB

  const int t     = threadIdx.x;
  const int lane  = t & 31;
  const int wv    = t >> 5;
  const int row0  = blockIdx.x * 128;    // row0+127 < MPAD always
  const int lrow  = lane & 15;           // row (A) / col (B) within 16
  const int kb    = (lane >> 4) * 8;     // K sub-base per ISA 16-bit layout
  const int nbase = wv * 32;

  const v8f vzero = {};
  v8f cacc[8][2];
  #pragma unroll
  for (int i = 0; i < 8; ++i)
    #pragma unroll
    for (int j = 0; j < 2; ++j) cacc[i][j] = vzero;

  auto issue = [&](int kk, int buf) {
    const unsigned aBase = (unsigned)(uintptr_t)&ldsA[buf][0];
    const unsigned bBase = (unsigned)(uintptr_t)&ldsB[buf][0];
    // A tile: 128 rows x 64B  = 512 x 16B chunks, 2 per thread
    #pragma unroll
    for (int i = 0; i < 2; ++i) {
      int idx = t + i * 256;
      int r = idx >> 2, c = idx & 3;                 // 4 chunks per row
      const char* g = (const char*)(Abf + (size_t)(row0 + r) * DMODEL + kk * 32) + c * 16;
      async_cp16(aBase + (unsigned)(r * 64 + c * 16), g);
    }
    // B tile: 256 cols x 64B = 1024 x 16B chunks, 4 per thread (contiguous in Wt)
    #pragma unroll
    for (int i = 0; i < 4; ++i) {
      int idx = t + i * 256;
      int n = idx >> 2, c = idx & 3;
      const char* g = (const char*)(Wt + (size_t)n * DMODEL + kk * 32) + c * 16;
      async_cp16(bBase + (unsigned)(n * 64 + c * 16), g);
    }
  };

  issue(0, 0);
  for (int kk = 0; kk < 8; ++kk) {
    asm volatile("s_wait_asynccnt 0" ::: "memory");  // my copies for buf kk&1 done
    __syncthreads();                                  // everyone's copies done
    if (kk < 7) issue(kk + 1, (kk + 1) & 1);          // overlap DMA with WMMA

    const __bf16* bufA = &ldsA[kk & 1][0];
    const __bf16* bufB = &ldsB[kk & 1][0];

    v16bf bfr[2];
    #pragma unroll
    for (int ct = 0; ct < 2; ++ct) {
      const __bf16* pb = &bufB[(nbase + ct * 16 + lrow) * 32 + kb];
      #pragma unroll
      for (int i = 0; i < 8; ++i) { bfr[ct][i] = pb[i]; bfr[ct][8 + i] = pb[16 + i]; }
    }
    #pragma unroll
    for (int rt = 0; rt < 8; ++rt) {
      const __bf16* pa = &bufA[(rt * 16 + lrow) * 32 + kb];
      v16bf afr;
      #pragma unroll
      for (int i = 0; i < 8; ++i) { afr[i] = pa[i]; afr[8 + i] = pa[16 + i]; }
      #pragma unroll
      for (int ct = 0; ct < 2; ++ct) {
        cacc[rt][ct] = __builtin_amdgcn_wmma_f32_16x16x32_bf16(
            false, afr, false, bfr[ct], (short)0, cacc[rt][ct], false, false);
      }
    }
  }

  // Epilogue: C/D layout => element v: row = (lane>=16)*8 + v, col = lane&15
  #pragma unroll
  for (int rt = 0; rt < 8; ++rt) {
    #pragma unroll
    for (int ct = 0; ct < 2; ++ct) {
      #pragma unroll
      for (int v = 0; v < 8; ++v) {
        int row = row0 + rt * 16 + (lane >> 4) * 8 + v;
        int col = nbase + ct * 16 + (lane & 15);
        if (row < M) {
          float val = cacc[rt][ct][v];
          if (use_bias) val += bias[col];
          if (use_act)  val = (val > 0.0f) ? val : slope * val;
          size_t o = (size_t)row * DMODEL + col;
          if (mode == 0) Cout[o] = val;
          else           Acc[o] += decay * val;
        }
      }
    }
  }
}

// ---------------------------------------------------------------------------
// Attention logits: as[n,h] = <hk[n,h,:], a_src[h,:]>, same for ad.
// ---------------------------------------------------------------------------
__global__ void alpha_kernel(const float* __restrict__ hk,
                             const float* __restrict__ asrc,
                             const float* __restrict__ adst,
                             float* __restrict__ as_, float* __restrict__ ad_)
{
  int t = blockIdx.x * blockDim.x + threadIdx.x;
  if (t >= NNODES * NHEADS) return;
  int n = t >> 3, hh = t & 7;
  const float* hp = hk + (size_t)n * DMODEL + hh * DHEAD;
  const float* ps = asrc + hh * DHEAD;
  const float* pd = adst + hh * DHEAD;
  float s1 = 0.0f, s2 = 0.0f;
  #pragma unroll
  for (int d = 0; d < DHEAD; ++d) { s1 += hp[d] * ps[d]; s2 += hp[d] * pd[d]; }
  as_[t] = s1; ad_[t] = s2;
}

__global__ void zero_kernel(float* __restrict__ p, int n)
{
  int t = blockIdx.x * blockDim.x + threadIdx.x;
  if (t < n) p[t] = 0.0f;
}

__global__ void init_ms_kernel(float* __restrict__ m_, float* __restrict__ s_)
{
  int t = blockIdx.x * blockDim.x + threadIdx.x;
  if (t < NNODES * NHEADS) { m_[t] = -3.0e38f; s_[t] = 0.0f; }
}

__global__ void init_gatout_kernel(float* __restrict__ go, const float* __restrict__ b)
{
  int t = blockIdx.x * blockDim.x + threadIdx.x;
  if (t < NNODES * DMODEL) go[t] = b[t & (DMODEL - 1)];
}

__device__ inline void atomicMaxF(float* addr, float val)
{
  unsigned int* ua = (unsigned int*)addr;
  unsigned int old = __float_as_uint(*addr);
  while (__uint_as_float(old) < val) {
    unsigned int assumed = old;
    old = atomicCAS(ua, assumed, __float_as_uint(val));
    if (old == assumed) break;
  }
}

__global__ void edge_max_kernel(const int* __restrict__ src, const int* __restrict__ dst,
                                const float* __restrict__ as_, const float* __restrict__ ad_,
                                float* __restrict__ m_)
{
  int t = blockIdx.x * blockDim.x + threadIdx.x;
  if (t >= NEDGES * NHEADS) return;
  int e = t >> 3, hh = t & 7;
  float v = as_[src[e] * NHEADS + hh] + ad_[dst[e] * NHEADS + hh];
  v = (v > 0.0f) ? v : 0.2f * v;
  atomicMaxF(&m_[dst[e] * NHEADS + hh], v);
}

__global__ void edge_expsum_kernel(const int* __restrict__ src, const int* __restrict__ dst,
                                   const float* __restrict__ as_, const float* __restrict__ ad_,
                                   const float* __restrict__ m_, float* __restrict__ s_,
                                   float* __restrict__ wE)
{
  int t = blockIdx.x * blockDim.x + threadIdx.x;
  if (t >= NEDGES * NHEADS) return;
  int e = t >> 3, hh = t & 7;
  int dN = dst[e];
  float v = as_[src[e] * NHEADS + hh] + ad_[dN * NHEADS + hh];
  v = (v > 0.0f) ? v : 0.2f * v;
  float w = expf(v - m_[dN * NHEADS + hh]);
  wE[t] = w;
  unsafeAtomicAdd(&s_[dN * NHEADS + hh], w);
}

// One wave per edge; lane d covers dim d of each head => 128B coalesced atomics
__global__ void edge_scatter_kernel(const int* __restrict__ src, const int* __restrict__ dst,
                                    const float* __restrict__ hk, const float* __restrict__ wE,
                                    const float* __restrict__ s_, float* __restrict__ go)
{
  int wid = (blockIdx.x * blockDim.x + threadIdx.x) >> 5;
  int lane = threadIdx.x & 31;
  if (wid >= NEDGES) return;
  int sN = src[wid], dN = dst[wid];
  const float* hs = hk + (size_t)sN * DMODEL;
  float* od = go + (size_t)dN * DMODEL;
  #pragma unroll
  for (int hh = 0; hh < NHEADS; ++hh) {
    float a = wE[wid * NHEADS + hh] / (s_[dN * NHEADS + hh] + 1e-16f);
    unsafeAtomicAdd(&od[hh * DHEAD + lane], hs[hh * DHEAD + lane] * a);
  }
}

// One wave per row: out = LN(acc)*scale + bias + resid
__global__ void ln_res_kernel(const float* __restrict__ acc, const float* __restrict__ resid,
                              const float* __restrict__ scale, const float* __restrict__ bias,
                              float* __restrict__ out)
{
  int wid = (blockIdx.x * blockDim.x + threadIdx.x) >> 5;
  int lane = threadIdx.x & 31;
  if (wid >= NNODES) return;
  const float* a = acc + (size_t)wid * DMODEL;
  float vals[8];
  float sum = 0.0f;
  #pragma unroll
  for (int j = 0; j < 8; ++j) { vals[j] = a[lane + j * 32]; sum += vals[j]; }
  #pragma unroll
  for (int off = 16; off >= 1; off >>= 1) sum += __shfl_xor(sum, off, 32);
  float mu = sum * (1.0f / DMODEL);
  float vs = 0.0f;
  #pragma unroll
  for (int j = 0; j < 8; ++j) { float d = vals[j] - mu; vs += d * d; }
  #pragma unroll
  for (int off = 16; off >= 1; off >>= 1) vs += __shfl_xor(vs, off, 32);
  float inv = rsqrtf(vs * (1.0f / DMODEL) + 1e-5f);
  #pragma unroll
  for (int j = 0; j < 8; ++j) {
    int c = lane + j * 32;
    out[(size_t)wid * DMODEL + c] =
        (vals[j] - mu) * inv * scale[c] + bias[c] + resid[(size_t)wid * DMODEL + c];
  }
}

// ---------------------------------------------------------------------------
extern "C" void kernel_launch(void* const* d_in, const int* in_sizes, int n_in,
                              void* d_out, int out_size, void* d_ws, size_t ws_size,
                              hipStream_t stream)
{
  const float* x      = (const float*)d_in[0];
  const int*   ei     = (const int*)  d_in[1];   // [3,2,E]
  const float* lin1_w = (const float*)d_in[2];
  const float* lin1_b = (const float*)d_in[3];
  const float* gat_w  = (const float*)d_in[4];   // [2,3,256,256]
  const float* att_s  = (const float*)d_in[5];   // [2,3,8,32]
  const float* att_d  = (const float*)d_in[6];
  const float* gat_b  = (const float*)d_in[7];   // [2,3,256]
  const float* dec_w  = (const float*)d_in[8];   // [2,3,256,256]
  const float* dec_b  = (const float*)d_in[9];   // [2,3,256]
  const float* ln_s   = (const float*)d_in[10];  // [2,256]
  const float* ln_b   = (const float*)d_in[11];
  float* out = (float*)d_out;

  // Workspace layout: ~272 MB total
  const size_t NT = (size_t)NNODES * DMODEL;
  float* ws  = (float*)d_ws;
  float* h   = ws;                                   // [N,256]
  float* acc = ws + NT;                              // [N,256]
  float* hk  = ws + 2 * NT;                          // [N,256]
  float* go  = ws + 3 * NT;                          // [N,256]
  float* as_ = ws + 4 * NT;                          // [N,8]
  float* ad_ = as_ + (size_t)NNODES * NHEADS;        // [N,8]
  float* m_  = ad_ + (size_t)NNODES * NHEADS;        // [N,8]
  float* s_  = m_  + (size_t)NNODES * NHEADS;        // [N,8]
  float* wE  = s_  + (size_t)NNODES * NHEADS;        // [E,8]
  __bf16* abf_h = (__bf16*)(wE + (size_t)NEDGES * NHEADS);   // [MPAD,256] bf16
  __bf16* abf_g = abf_h + (size_t)MPAD * DMODEL;             // [MPAD,256] bf16
  __bf16* wbf   = abf_g + (size_t)MPAD * DMODEL;             // [256,256]  bf16 (transposed)

  const dim3 blk(256);
  const int gGemm = MPAD / 128;                         // 391 blocks
  const int gCvtA = (MPAD * DMODEL + 255) / 256;
  const int gCvtW = (DMODEL * DMODEL + 255) / 256;
  const int gNH   = (NNODES * NHEADS + 255) / 256;
  const int gND   = ((int)NT + 255) / 256;
  const int gEH   = (NEDGES * NHEADS + 255) / 256;
  const int gEW   = (NEDGES * 32 + 255) / 256;          // wave per edge
  const int gNW   = (NNODES * 32 + 255) / 256;          // wave per row
  const float decay[3] = {1.0f, 0.60653065971263342f, 0.36787944117144233f};

  // h = leaky_relu(x @ lin1_w + lin1_b, 0.01)
  cvtA_kernel<<<gCvtA, blk, 0, stream>>>(x, abf_h, NNODES);
  cvtW_kernel<<<gCvtW, blk, 0, stream>>>(lin1_w, wbf);
  gemm_wmma_async<<<gGemm, blk, 0, stream>>>(abf_h, wbf, lin1_b, h, nullptr,
                                             NNODES, 0.01f, 1, 1, 0, 0.0f);

  for (int l = 0; l < 2; ++l) {
    cvtA_kernel<<<gCvtA, blk, 0, stream>>>(h, abf_h, NNODES);  // h fixed within layer
    zero_kernel<<<gND, blk, 0, stream>>>(acc, (int)NT);
    for (int k = 0; k < 3; ++k) {
      const size_t lk = (size_t)(l * 3 + k);
      const float* Wg  = gat_w + lk * DMODEL * DMODEL;
      const float* Wd  = dec_w + lk * DMODEL * DMODEL;
      const float* bd  = dec_b + lk * DMODEL;
      const float* bg  = gat_b + lk * DMODEL;
      const float* asv = att_s + lk * (NHEADS * DHEAD);
      const float* adv = att_d + lk * (NHEADS * DHEAD);
      const int* src = ei + ((size_t)k * 2 + 0) * NEDGES;
      const int* dst = ei + ((size_t)k * 2 + 1) * NEDGES;

      // hk = h @ gat_w[l,k]
      cvtW_kernel<<<gCvtW, blk, 0, stream>>>(Wg, wbf);
      gemm_wmma_async<<<gGemm, blk, 0, stream>>>(abf_h, wbf, nullptr, hk, nullptr,
                                                 NNODES, 0.0f, 0, 0, 0, 0.0f);
      alpha_kernel<<<gNH, blk, 0, stream>>>(hk, asv, adv, as_, ad_);
      init_ms_kernel<<<gNH, blk, 0, stream>>>(m_, s_);
      init_gatout_kernel<<<gND, blk, 0, stream>>>(go, bg);  // go starts at gat bias
      edge_max_kernel<<<gEH, blk, 0, stream>>>(src, dst, as_, ad_, m_);
      edge_expsum_kernel<<<gEH, blk, 0, stream>>>(src, dst, as_, ad_, m_, s_, wE);
      edge_scatter_kernel<<<gEW, blk, 0, stream>>>(src, dst, hk, wE, s_, go);
      // acc += decay[k] * leaky_relu(go @ dec_w + dec_b, 0.01)
      cvtA_kernel<<<gCvtA, blk, 0, stream>>>(go, abf_g, NNODES);
      cvtW_kernel<<<gCvtW, blk, 0, stream>>>(Wd, wbf);
      gemm_wmma_async<<<gGemm, blk, 0, stream>>>(abf_g, wbf, bd, nullptr, acc,
                                                 NNODES, 0.01f, 1, 1, 1, decay[k]);
    }
    float* dstbuf = (l == 0) ? h : out;
    ln_res_kernel<<<gNW, blk, 0, stream>>>(acc, h, ln_s + l * DMODEL,
                                           ln_b + l * DMODEL, dstbuf);
  }
}
